// PFormerAttention_33715493273773
// MI455X (gfx1250) — matrix operand order repaired and checked by reading, not compiled, not run on hardware
//
#include <hip/hip_runtime.h>

// ---------------- types / helpers ----------------
typedef __attribute__((ext_vector_type(16))) __bf16 v16bf;
typedef __attribute__((ext_vector_type(8)))  __bf16 v8bf;
typedef __attribute__((ext_vector_type(8)))  float  v8f;

union V16 { v16bf v; v8bf h[2]; __bf16 e[16]; float4 f4[2]; };
union V8F { v8f v; float f[8]; };

#define DEV __device__ __forceinline__

DEV __bf16 f2bf(float f) {
  union { float f; unsigned u; } a; a.f = f;
  unsigned r = a.u + 0x7FFFu + ((a.u >> 16) & 1u);   // round-to-nearest-even
  union { unsigned short s; __bf16 b; } c; c.s = (unsigned short)(r >> 16);
  return c.b;
}

DEV v8f v8f_zero() {
  V8F z;
#pragma unroll
  for (int i = 0; i < 8; ++i) z.f[i] = 0.f;
  return z.v;
}

#define WMMA_BF16(A, B, C) \
  __builtin_amdgcn_wmma_f32_16x16x32_bf16(false, (A), false, (B), (short)0, (C), false, false)

// ---------------- prep kernels ----------------
__global__ void k_cvt_x(const float* __restrict__ x, __bf16* __restrict__ xb) {
  int t = blockIdx.x * 256 + threadIdx.x;
  xb[t] = f2bf(x[t]);
}

// wbT[co][ci] = w_qkv[ci][co]   (w_qkv is [512,1536] row-major)
__global__ void k_wT(const float* __restrict__ w, __bf16* __restrict__ wbT) {
  int t = blockIdx.x * 256 + threadIdx.x;
  int co = t >> 9, ci = t & 511;
  wbT[t] = f2bf(w[ci * 1536 + co]);
}

// u[row][f] = pos[row]·w1[:,f]   (rows = B*N = 4096, f = 0..63)
__global__ void k_u(const float* __restrict__ pos, const float* __restrict__ w1,
                    float* __restrict__ u) {
  int t = blockIdx.x * 256 + threadIdx.x;
  int row = t >> 6, f = t & 63;
  u[t] = pos[row * 3 + 0] * w1[f] + pos[row * 3 + 1] * w1[64 + f] +
         pos[row * 3 + 2] * w1[128 + f];
}

// ---------------- QKV GEMM: [4096,512] @ [512,1536] + b ----------------
__global__ __launch_bounds__(256) void k_qkv(const __bf16* __restrict__ xb,
                                             const __bf16* __restrict__ wbT,
                                             const float* __restrict__ bq,
                                             __bf16* __restrict__ q,
                                             __bf16* __restrict__ k,
                                             __bf16* __restrict__ vT) {
  __shared__ __align__(16) __bf16 sA[128 * 32];
  __shared__ __align__(16) __bf16 sB[128 * 32];
  const int tid = threadIdx.x;
  const int w = tid >> 5, lane = tid & 31, lm = lane & 15;
  const bool hi = lane >= 16;
  const int rowBase = blockIdx.y * 128, colBase = blockIdx.x * 128;

  v8f acc[8];
#pragma unroll
  for (int s = 0; s < 8; ++s) acc[s] = v8f_zero();

  for (int kk = 0; kk < 512; kk += 32) {
    __syncthreads();
    {
      int r = tid >> 1, hf = (tid & 1) * 16;
      const float4* srcA = (const float4*)(xb + (size_t)(rowBase + r) * 512 + kk + hf);
      float4* dstA = (float4*)(sA + r * 32 + hf);
      dstA[0] = srcA[0]; dstA[1] = srcA[1];
      const float4* srcB = (const float4*)(wbT + (size_t)(colBase + r) * 512 + kk + hf);
      float4* dstB = (float4*)(sB + r * 32 + hf);
      dstB[0] = srcB[0]; dstB[1] = srcB[1];
    }
    __syncthreads();
    V16 a;
    const __bf16* ap = sA + (w * 16 + lm) * 32;
    a.f4[0] = *(const float4*)(ap + (hi ? 8 : 0));
    a.f4[1] = *(const float4*)(ap + (hi ? 24 : 16));
#pragma unroll
    for (int s = 0; s < 8; ++s) {
      V16 bb;
      const __bf16* bp = sB + (s * 16 + lm) * 32 + (hi ? 8 : 0);
      bb.f4[0] = *(const float4*)bp;
      bb.f4[1] = *(const float4*)(bp + 16);
      acc[s] = WMMA_BF16(a.v, bb.v, acc[s]);
    }
  }

  // epilogue: +bias, scatter into q / k / vT (bf16)
#pragma unroll
  for (int s = 0; s < 8; ++s) {
    int col = colBase + s * 16 + lm;
    float bias = bq[col];
    int which = col >> 9;            // 0=q 1=k 2=v (uniform per subtile)
    int rem = col & 511;
    int h = rem >> 6, d = rem & 63;
    V8F av; av.v = acc[s];
#pragma unroll
    for (int r = 0; r < 8; ++r) {
      int row = rowBase + w * 16 + r + (hi ? 8 : 0);
      int b = row >> 10, n = row & 1023;
      __bf16 bv = f2bf(av.f[r] + bias);
      if (which == 0)      q[((size_t)(b * 8 + h) * 1024 + n) * 64 + d] = bv;
      else if (which == 1) k[((size_t)(b * 8 + h) * 1024 + n) * 64 + d] = bv;
      else                 vT[((size_t)(b * 8 + h) * 64 + d) * 1024 + n] = bv;
    }
  }
}

// ---------------- fused attention + pair-MLP bias ----------------
__global__ __launch_bounds__(256) void k_attn(
    const __bf16* __restrict__ qws, const __bf16* __restrict__ kws,
    const __bf16* __restrict__ vTws, const float* __restrict__ u,
    const float* __restrict__ pos, const float* __restrict__ b1g,
    const float* __restrict__ w2g, const float* __restrict__ b2g,
    float* __restrict__ out) {
  __shared__ __align__(16) float  s_un[16][64];
  __shared__ __align__(16) float  s_um[16][64];
  __shared__ float  s_posn[16][4];
  __shared__ float  s_posm[16][4];
  __shared__ float  s_b1[64];
  __shared__ __align__(16) __bf16 s_h1[16][16][64];   // [n][m][feat]
  __shared__ __align__(16) float  s_bias[16][16][8];  // [n][m][h]
  __shared__ __align__(16) __bf16 s_p[8][16][16];     // per-wave P transpose

  const int tid = threadIdx.x;
  const int w = tid >> 5, lane = tid & 31, lm = lane & 15;   // w == head
  const bool hi = lane >= 16;
  const int b = blockIdx.y, n0 = blockIdx.x * 16;
  const float4 z4 = {0.f, 0.f, 0.f, 0.f};

  // stage u_n / pos_n / b1
  { int r = tid >> 4, c4 = (tid & 15) * 4;
    *(float4*)&s_un[r][c4] = *(const float4*)&u[((size_t)(b * 1024 + n0 + r)) * 64 + c4]; }
  if (tid < 16) {
    size_t p = (size_t)(b * 1024 + n0 + tid) * 3;
    s_posn[tid][0] = pos[p + 0]; s_posn[tid][1] = pos[p + 1]; s_posn[tid][2] = pos[p + 2];
  }
  if (tid < 64) s_b1[tid] = b1g[tid];

  // q fragments (rows n0..n0+15, K = head-dim 0..63), head = w
  const __bf16* qb = qws + ((size_t)(b * 8 + w) * 1024 + n0) * 64;
  V16 qa0, qa1;
  { const __bf16* qp = qb + lm * 64;
    qa0.f4[0] = *(const float4*)(qp + (hi ? 8 : 0));
    qa0.f4[1] = *(const float4*)(qp + (hi ? 24 : 16));
    qa1.f4[0] = *(const float4*)(qp + 32 + (hi ? 8 : 0));
    qa1.f4[1] = *(const float4*)(qp + 32 + (hi ? 24 : 16)); }

  // w2 B-fragments: K=64 feats x N=16 (heads 8..15 zero)
  V16 w2f0, w2f1;
#pragma unroll
  for (int e2 = 0; e2 < 16; ++e2) {
    int k0 = (hi ? 8 : 0) + (e2 < 8 ? e2 : e2 + 8);
    int k1 = 32 + (hi ? 8 : 0) + (e2 < 8 ? e2 : e2 + 8);
    float v0 = (lm < 8) ? w2g[k0 * 8 + lm] : 0.f;
    float v1 = (lm < 8) ? w2g[k1 * 8 + lm] : 0.f;
    w2f0.e[e2] = f2bf(v0);
    w2f1.e[e2] = f2bf(v1);
  }

  v8f o[4];
#pragma unroll
  for (int t4 = 0; t4 < 4; ++t4) o[t4] = v8f_zero();
  float m_run[8], l_run[8];
#pragma unroll
  for (int r = 0; r < 8; ++r) { m_run[r] = -1e30f; l_run[r] = 0.f; }

  for (int m0 = 0; m0 < 1024; m0 += 16) {
    __syncthreads();
    { int r = tid >> 4, c4 = (tid & 15) * 4;
      *(float4*)&s_um[r][c4] = *(const float4*)&u[((size_t)(b * 1024 + m0 + r)) * 64 + c4]; }
    if (tid < 16) {
      size_t p = (size_t)(b * 1024 + m0 + tid) * 3;
      s_posm[tid][0] = pos[p + 0]; s_posm[tid][1] = pos[p + 1]; s_posm[tid][2] = pos[p + 2];
    }
    __syncthreads();

    // h1[i][j][:] = relu((u_n - u_m) / ||pos_n - pos_m|| + b1)
    { int i = tid >> 4, j = tid & 15;
      float dx = s_posn[i][0] - s_posm[j][0];
      float dy = s_posn[i][1] - s_posm[j][1];
      float dz = s_posn[i][2] - s_posm[j][2];
      float rn = 1.f / (sqrtf(dx * dx + dy * dy + dz * dz) + 1e-6f);
#pragma unroll 8
      for (int f = 0; f < 64; ++f) {
        float hv = (s_un[i][f] - s_um[j][f]) * rn + s_b1[f];
        s_h1[i][j][f] = f2bf(fmaxf(hv, 0.f));
      }
    }
    __syncthreads();

    // bias layer-2 via WMMA: per n-row, [16m x 64k] @ [64k x 16h]
#pragma unroll
    for (int ii = 0; ii < 2; ++ii) {
      int i = 2 * w + ii;
      V16 a0, a1;
      const __bf16* hp = &s_h1[i][lm][0];
      a0.f4[0] = *(const float4*)(hp + (hi ? 8 : 0));
      a0.f4[1] = *(const float4*)(hp + (hi ? 24 : 16));
      a1.f4[0] = *(const float4*)(hp + 32 + (hi ? 8 : 0));
      a1.f4[1] = *(const float4*)(hp + 32 + (hi ? 24 : 16));
      v8f dacc = v8f_zero();
      dacc = WMMA_BF16(a0.v, w2f0.v, dacc);
      dacc = WMMA_BF16(a1.v, w2f1.v, dacc);
      V8F dv; dv.v = dacc;
      if (lm < 8) {
        float bb2 = b2g[lm];
#pragma unroll
        for (int r = 0; r < 8; ++r)
          s_bias[i][r + (hi ? 8 : 0)][lm] = dv.f[r] + bb2;
      }
    }
    __syncthreads();

    // S = q @ k^T  (2 WMMAs over K=64)
    V16 kb0, kb1;
    const __bf16* kb = kws + ((size_t)(b * 8 + w) * 1024 + m0) * 64;
    { const __bf16* kp = kb + lm * 64 + (hi ? 8 : 0);
      kb0.f4[0] = *(const float4*)kp;
      kb0.f4[1] = *(const float4*)(kp + 16);
      const __bf16* kp1 = kb + lm * 64 + 32 + (hi ? 8 : 0);
      kb1.f4[0] = *(const float4*)kp1;
      kb1.f4[1] = *(const float4*)(kp1 + 16); }
    v8f sacc = v8f_zero();
    sacc = WMMA_BF16(qa0.v, kb0.v, sacc);
    sacc = WMMA_BF16(qa1.v, kb1.v, sacc);
    V8F sv; sv.v = sacc;

    // online softmax (rows live across a 16-lane half)
    float pp[8], alpha[8];
#pragma unroll
    for (int r = 0; r < 8; ++r) {
      int n = r + (hi ? 8 : 0);
      float x = sv.f[r] * 0.125f + s_bias[n][lm][w];
      float rmax = x;
      rmax = fmaxf(rmax, __shfl_xor(rmax, 1, 32));
      rmax = fmaxf(rmax, __shfl_xor(rmax, 2, 32));
      rmax = fmaxf(rmax, __shfl_xor(rmax, 4, 32));
      rmax = fmaxf(rmax, __shfl_xor(rmax, 8, 32));
      float mn = fmaxf(m_run[r], rmax);
      float pe = __expf(x - mn);
      float rs = pe;
      rs += __shfl_xor(rs, 1, 32);
      rs += __shfl_xor(rs, 2, 32);
      rs += __shfl_xor(rs, 4, 32);
      rs += __shfl_xor(rs, 8, 32);
      alpha[r] = __expf(m_run[r] - mn);
      l_run[r] = l_run[r] * alpha[r] + rs;
      m_run[r] = mn;
      pp[r] = pe;
    }
#pragma unroll
    for (int t4 = 0; t4 < 4; ++t4) {
      V8F ov; ov.v = o[t4];
#pragma unroll
      for (int r = 0; r < 8; ++r) ov.f[r] *= alpha[r];
      o[t4] = ov.v;
    }

    // transpose P through per-wave LDS scratch (same-wave DS ops are in order)
#pragma unroll
    for (int r = 0; r < 8; ++r) s_p[w][r + (hi ? 8 : 0)][lm] = f2bf(pp[r]);
    asm volatile("s_wait_dscnt 0" ::: "memory");
    V16 pa;
    { const __bf16* ppt = &s_p[w][lm][hi ? 8 : 0];
      pa.f4[0] = *(const float4*)ppt;
      pa.f4[1] = z4; }                       // K 16..31 padded with zeros

    // O += P @ V  (vT rows are contiguous along m)
    const __bf16* vbse = vTws + ((size_t)(b * 8 + w) * 64) * 1024 + m0;
#pragma unroll
    for (int t4 = 0; t4 < 4; ++t4) {
      V16 vb;
      const __bf16* vp = vbse + (size_t)(t4 * 16 + lm) * 1024;
      vb.f4[0] = *(const float4*)(vp + (hi ? 8 : 0));
      vb.f4[1] = z4;                              // padded K slots
      o[t4] = WMMA_BF16(pa.v, vb.v, o[t4]);
    }
  }

  // normalize and write out[b][n][h*64+d] (fp32)
#pragma unroll
  for (int t4 = 0; t4 < 4; ++t4) {
    V8F ov; ov.v = o[t4];
#pragma unroll
    for (int r = 0; r < 8; ++r) {
      int n = n0 + r + (hi ? 8 : 0);
      int d = t4 * 16 + lm;
      out[((size_t)(b * 1024 + n)) * 512 + w * 64 + d] = ov.f[r] / l_run[r];
    }
  }
}

// ---------------- host launch ----------------
extern "C" void kernel_launch(void* const* d_in, const int* in_sizes, int n_in,
                              void* d_out, int out_size, void* d_ws, size_t ws_size,
                              hipStream_t stream) {
  (void)in_sizes; (void)n_in; (void)out_size; (void)ws_size;
  const float* x    = (const float*)d_in[0];
  const float* pos  = (const float*)d_in[1];
  const float* wqkv = (const float*)d_in[2];
  const float* bqkv = (const float*)d_in[3];
  const float* w1   = (const float*)d_in[4];
  const float* b1   = (const float*)d_in[5];
  const float* w2   = (const float*)d_in[6];
  const float* b2   = (const float*)d_in[7];
  float* out = (float*)d_out;

  char* ws = (char*)d_ws;
  __bf16* xb  = (__bf16*)(ws);                 // 4096*512  bf16 = 4 MB
  __bf16* wbT = (__bf16*)(ws + 4194304);       // 1536*512  bf16 = 1.5 MB
  __bf16* q   = (__bf16*)(ws + 5767168);       // [B,H,N,D] bf16 = 4 MB
  __bf16* k   = (__bf16*)(ws + 9961472);       // [B,H,N,D] bf16 = 4 MB
  __bf16* vT  = (__bf16*)(ws + 14155776);      // [B,H,D,N] bf16 = 4 MB
  float*  u   = (float*)(ws + 18350080);       // [B*N,64]  f32  = 1 MB

  k_cvt_x<<<dim3(8192), dim3(256), 0, stream>>>(x, xb);
  k_wT<<<dim3(3072), dim3(256), 0, stream>>>(wqkv, wbT);
  k_u<<<dim3(1024), dim3(256), 0, stream>>>(pos, w1, u);
  k_qkv<<<dim3(12, 32), dim3(256), 0, stream>>>(xb, wbT, bqkv, q, k, vT);
  k_attn<<<dim3(64, 4), dim3(256), 0, stream>>>(q, k, vT, u, pos, b1, w2, b2, out);
}
